// IntActWeight_21311627723417
// MI455X (gfx1250) — 8-bit matrix operand order repaired and checked by reading, compiled, not run on hardware
//
#include <hip/hip_runtime.h>

typedef __attribute__((ext_vector_type(8))) int v8i;

#define TILE_M 128
#define TILE_N 128
#define TILE_K 64
#define LDS_STRIDE 80
#define BUF_BYTES (TILE_M * LDS_STRIDE)
#define NBUF 3

__device__ __forceinline__ void async_load_b128(unsigned lds_off, const void* gaddr) {
    asm volatile("global_load_async_to_lds_b128 %0, %1, off"
                 :: "v"(lds_off), "v"(gaddr)
                 : "memory");
}

__device__ __forceinline__ void wait_async_le0() {
#if __has_builtin(__builtin_amdgcn_s_wait_asynccnt)
    __builtin_amdgcn_s_wait_asynccnt(0);
#else
    asm volatile("s_wait_asynccnt 0" ::: "memory");
#endif
}

__device__ __forceinline__ void wait_async_le4() {
#if __has_builtin(__builtin_amdgcn_s_wait_asynccnt)
    __builtin_amdgcn_s_wait_asynccnt(4);
#else
    asm volatile("s_wait_asynccnt 4" ::: "memory");
#endif
}

__global__ void pack_x_s8(const int* __restrict__ in, signed char* __restrict__ out, int n4) {
    int i = blockIdx.x * blockDim.x + threadIdx.x;
    if (i < n4) {
        int4 v = ((const int4*)in)[i];
        char4 p;
        p.x = (signed char)v.x;
        p.y = (signed char)v.y;
        p.z = (signed char)v.z;
        p.w = (signed char)v.w;
        ((char4*)out)[i] = p;
    }
}

__global__ void pack_y_t_s8(const int* __restrict__ y, signed char* __restrict__ yt,
                            int K, int N) {
    __shared__ signed char tile[64][68];
    const int k0 = blockIdx.y * 64;
    const int n0 = blockIdx.x * 64;
    const int tid = threadIdx.x;
    const int c = tid & 63;
    const int rbase = (tid >> 6) * 16;
#pragma unroll
    for (int i = 0; i < 16; ++i) {
        int r = rbase + i;
        tile[c][r] = (signed char)y[(size_t)(k0 + r) * N + n0 + c];
    }
    __syncthreads();
#pragma unroll
    for (int i = 0; i < 4; ++i) {
        int idx = tid + i * 256;
        int row = idx >> 4;
        int ch  = idx & 15;
        char4 p;
        p.x = tile[row][ch * 4 + 0];
        p.y = tile[row][ch * 4 + 1];
        p.z = tile[row][ch * 4 + 2];
        p.w = tile[row][ch * 4 + 3];
        *(char4*)&yt[(size_t)(n0 + row) * K + k0 + ch * 4] = p;
    }
}

__global__ __launch_bounds__(256) void gemm_i8_wmma(
    const signed char* __restrict__ A, const signed char* __restrict__ Bt,
    float* __restrict__ out, const float* __restrict__ scale_p,
    int M, int N, int K) {

    __shared__ __align__(16) signed char lA[NBUF][BUF_BYTES];
    __shared__ __align__(16) signed char lB[NBUF][BUF_BYTES];

    const int tid  = threadIdx.x;
    const int lane = tid & 31;
    const int wave = tid >> 5;
    const int wm   = wave >> 2;
    const int wn   = wave & 3;

    const int m0 = blockIdx.y * TILE_M;
    const int n0 = blockIdx.x * TILE_N;

    const int chunk = tid & 3;
    const int row   = tid >> 2;
    const signed char* gA0 = A  + (size_t)(m0 + row)      * K + chunk * 16;
    const signed char* gA1 = A  + (size_t)(m0 + row + 64) * K + chunk * 16;
    const signed char* gB0 = Bt + (size_t)(n0 + row)      * K + chunk * 16;
    const signed char* gB1 = Bt + (size_t)(n0 + row + 64) * K + chunk * 16;

    const unsigned stA0 = (unsigned)(uintptr_t)&lA[0][(size_t)row        * LDS_STRIDE + chunk * 16];
    const unsigned stA1 = (unsigned)(uintptr_t)&lA[0][(size_t)(row + 64) * LDS_STRIDE + chunk * 16];
    const unsigned stB0 = (unsigned)(uintptr_t)&lB[0][(size_t)row        * LDS_STRIDE + chunk * 16];
    const unsigned stB1 = (unsigned)(uintptr_t)&lB[0][(size_t)(row + 64) * LDS_STRIDE + chunk * 16];

    v8i acc[4][2];
#pragma unroll
    for (int mi = 0; mi < 4; ++mi)
#pragma unroll
        for (int ni = 0; ni < 2; ++ni)
            acc[mi][ni] = (v8i){0, 0, 0, 0, 0, 0, 0, 0};

    const int l15 = lane & 15;
    const int hi  = lane >> 4;
    const size_t aFragOff = (size_t)(wm * 64 + l15) * LDS_STRIDE + hi * 16;
    const size_t bFragOff = (size_t)(wn * 32 + l15) * LDS_STRIDE + hi * 16;

    const int KT = K / TILE_K;

    async_load_b128(stA0, gA0);
    async_load_b128(stA1, gA1);
    async_load_b128(stB0, gB0);
    async_load_b128(stB1, gB1);
    {
        const unsigned bo = (unsigned)BUF_BYTES;
        async_load_b128(stA0 + bo, gA0 + TILE_K);
        async_load_b128(stA1 + bo, gA1 + TILE_K);
        async_load_b128(stB0 + bo, gB0 + TILE_K);
        async_load_b128(stB1 + bo, gB1 + TILE_K);
    }
    wait_async_le4();
    __syncthreads();

    for (int kt = 0; kt < KT; ++kt) {
        const int buf = kt % NBUF;

        if (kt + 2 < KT) {
            const int goff = (kt + 2) * TILE_K;
            const unsigned boff = (unsigned)(((kt + 2) % NBUF) * BUF_BYTES);
            async_load_b128(stA0 + boff, gA0 + goff);
            async_load_b128(stA1 + boff, gA1 + goff);
            async_load_b128(stB0 + boff, gB0 + goff);
            async_load_b128(stB1 + boff, gB1 + goff);
        }

        const signed char* laBase = &lA[buf][0] + aFragOff;
        const signed char* lbBase = &lB[buf][0] + bFragOff;

        v8i fa[4];
#pragma unroll
        for (int mi = 0; mi < 4; ++mi) {
            const signed char* p = laBase + (size_t)(mi * 16) * LDS_STRIDE;
            int2 d0 = *(const int2*)(p);
            int2 d1 = *(const int2*)(p + 8);
            int2 d2 = *(const int2*)(p + 32);
            int2 d3 = *(const int2*)(p + 40);
            v8i f;
            f[0] = d0.x; f[1] = d0.y;
            f[2] = d1.x; f[3] = d1.y;
            f[4] = d2.x; f[5] = d2.y;
            f[6] = d3.x; f[7] = d3.y;
            fa[mi] = f;
        }
        v8i fb[2];
#pragma unroll
        for (int ni = 0; ni < 2; ++ni) {
            const signed char* p = lbBase + (size_t)(ni * 16) * LDS_STRIDE;
            int4 e0 = *(const int4*)(p);
            int4 e1 = *(const int4*)(p + 32);
            v8i f;
            f[0] = e0.x; f[1] = e0.y; f[2] = e0.z; f[3] = e0.w;
            f[4] = e1.x; f[5] = e1.y; f[6] = e1.z; f[7] = e1.w;
            fb[ni] = f;
        }

#pragma unroll
        for (int mi = 0; mi < 4; ++mi)
#pragma unroll
            for (int ni = 0; ni < 2; ++ni)
                acc[mi][ni] = __builtin_amdgcn_wmma_i32_16x16x64_iu8(
                     true, fa[mi],  true, fb[ni],
                    acc[mi][ni],  false,  false);

        if (kt + 1 < KT) {
            if (kt + 2 < KT) wait_async_le4();
            else             wait_async_le0();
            __syncthreads();
        }
    }

    const float s = *scale_p;
#pragma unroll
    for (int mi = 0; mi < 4; ++mi) {
#pragma unroll
        for (int ni = 0; ni < 2; ++ni) {
            const int mrow = m0 + wm * 64 + mi * 16 + hi * 8;
            const int ncol = n0 + wn * 32 + ni * 16 + l15;
#pragma unroll
            for (int r = 0; r < 8; ++r) {
                float v = (float)acc[mi][ni][r] * s;
                __builtin_nontemporal_store(v, &out[(size_t)(mrow + r) * N + ncol]);
            }
        }
    }
}

extern "C" void kernel_launch(void* const* d_in, const int* in_sizes, int n_in,
                              void* d_out, int out_size, void* d_ws, size_t ws_size,
                              hipStream_t stream) {
    (void)in_sizes; (void)n_in; (void)out_size; (void)ws_size;

    const int Bb = 4, S = 4096, K = 4096, N = 4096;
    const int M = Bb * S;

    const int*   x     = (const int*)d_in[0];
    const int*   y     = (const int*)d_in[1];
    const float* scale = (const float*)d_in[2];
    float*       out   = (float*)d_out;

    signed char* x8 = (signed char*)d_ws;
    signed char* yt = x8 + (size_t)M * K;

    const int n4 = (M / 4) * K;
    pack_x_s8<<<(n4 + 255) / 256, 256, 0, stream>>>(x, x8, n4);

    dim3 gy(N / 64, K / 64);
    pack_y_t_s8<<<gy, 256, 0, stream>>>(y, yt, K, N);

    dim3 gg(N / TILE_N, M / TILE_M);
    gemm_i8_wmma<<<gg, 256, 0, stream>>>(x8, yt, out, scale, M, N, K);
}
